// CausalSelfAttention_26156350832914
// MI455X (gfx1250) — matrix operand order repaired and checked by reading, not compiled, not run on hardware
//
#include <hip/hip_runtime.h>
#include <hip/hip_bf16.h>
#include <math.h>

// ---------------------------------------------------------------------------
// CausalSelfAttention for MI455X (gfx1250, wave32, WMMA f16->f32)
// B=4, S=2048, D=1024, H=16, hd=64
// Pipeline: f32->f16 | QKV GEMM (32x64/wave WMMA) | flash attention
//           (64-key chunks, online softmax) | out-proj GEMM + residual | LN
// ---------------------------------------------------------------------------

#define B_ 4
#define S_ 2048
#define D_ 1024
#define H_ 16
#define HD_ 64
#define M_TOK (B_ * S_)          // 8192 tokens
#define N_QKV (3 * D_)           // 3072
#define QK_SCALE 0.125f          // 1/sqrt(64)

typedef __attribute__((ext_vector_type(16))) _Float16 v16h;
typedef __attribute__((ext_vector_type(8)))  _Float16 v8h;
typedef __attribute__((ext_vector_type(8)))  float    v8f;

union AFrag { v16h v; v8h h[2]; };

// ---- WMMA fragment loaders (contiguous k = 16*khalf + i, same in A and B; wave32) -------------------
// A (16x32 f16, MxK): lane m=l&15, khalf=l>>4.
//   halfs[0..7]  = A[m][khalf*16 + 0..7]
//   halfs[8..15] = A[m][khalf*16 + 8 + 0..7]
__device__ __forceinline__ v16h load_a16x32(const _Float16* base, int ldk, int lane) {
    const int m = lane & 15, kh = lane >> 4;
    const _Float16* p = base + (size_t)m * ldk + kh * 16;
    AFrag f;
    f.h[0] = *(const v8h*)(p);
    f.h[1] = *(const v8h*)(p + 8);
    return f.v;
}

// B (32x16 f16, KxN), source stored as src[n][k] row-major (contiguous in k).
// lane n=l&15, khalf=l>>4: halfs[i] = B[khalf*16 + i][n] = src[n][khalf*16+i]
__device__ __forceinline__ v16h load_b32x16(const _Float16* base, int ldk, int lane) {
    const int n = lane & 15, kh = lane >> 4;
    return *(const v16h*)(base + (size_t)n * ldk + kh * 16);
}

__device__ __forceinline__ v8f wmma_f16(v16h a, v16h b, v8f c) {
    return __builtin_amdgcn_wmma_f32_16x16x32_f16(false, a, false, b, (short)0, c,
                                                  false, false);
}

// ---- 32x64 register-tile WMMA GEMM core (K = D_ = 1024) -------------------
// One wave: 2 M-subtiles x 4 N-subtiles, 8x v8f accumulators.
// All 6 fragments of a K-step are loaded up front (one load clause), then
// 8 WMMAs issue against them; A frag reused 4x, B frag reused 2x.
__device__ __forceinline__ void gemm_core_32x64(const _Float16* __restrict__ aBase,
                                                const _Float16* __restrict__ bBase,
                                                int lane, v8f c[2][4]) {
    #pragma unroll 2
    for (int k0 = 0; k0 < D_; k0 += 32) {
        // pull the next K-slab toward the WGP caches (WGP-scope prefetch)
        if (k0 + 64 < D_) {
            __builtin_prefetch(aBase + (size_t)(lane & 15) * D_ + k0 + 64, 0, 3);
            __builtin_prefetch(bBase + (size_t)(lane & 15) * D_ + k0 + 64, 0, 3);
        }
        const v16h a0 = load_a16x32(aBase + k0,                   D_, lane);
        const v16h a1 = load_a16x32(aBase + (size_t)16 * D_ + k0, D_, lane);
        v16h b[4];
        #pragma unroll
        for (int j = 0; j < 4; ++j)
            b[j] = load_b32x16(bBase + (size_t)(j * 16) * D_ + k0, D_, lane);
        #pragma unroll
        for (int j = 0; j < 4; ++j) {
            c[0][j] = wmma_f16(a0, b[j], c[0][j]);
            c[1][j] = wmma_f16(a1, b[j], c[1][j]);
        }
    }
}

// ---------------------------------------------------------------------------
// Kernel 0: fp32 -> fp16 conversion
// ---------------------------------------------------------------------------
__global__ void k_f32_to_f16(const float* __restrict__ in, _Float16* __restrict__ out, int n) {
    int i = blockIdx.x * blockDim.x + threadIdx.x;
    if (i < n) out[i] = (_Float16)in[i];
}

// ---------------------------------------------------------------------------
// Kernel 1: QKV GEMM  qkv[t][n] = sum_k xh[t][k] * W[n][k] + b[n]
// Scatter: q -> [B,H,S,hd] (scaled), k -> [B,H,S,hd], v -> [B,H,hd,S] (transposed)
// Block = 8 waves as 2(M) x 4(N); block tile 64 x 256.
// ---------------------------------------------------------------------------
__global__ void k_qkv_gemm(const _Float16* __restrict__ xh,
                           const _Float16* __restrict__ wh,
                           const float* __restrict__ bias,
                           _Float16* __restrict__ qbuf,
                           _Float16* __restrict__ kbuf,
                           _Float16* __restrict__ vtbuf) {
    const int lane = threadIdx.x & 31;
    const int wave = threadIdx.x >> 5;
    const int m0 = blockIdx.x * 64 + (wave >> 2) * 32;   // token tile base (32 rows)
    const int n0 = blockIdx.y * 256 + (wave & 3) * 64;   // feature tile base (64 cols)

    v8f c[2][4] = {};
    gemm_core_32x64(xh + (size_t)m0 * D_, wh + (size_t)n0 * D_, lane, c);

    const int mh = (lane >> 4) * 8;
    #pragma unroll
    for (int j = 0; j < 4; ++j) {
        const int n  = n0 + j * 16 + (lane & 15);
        const float bv = bias[n];
        #pragma unroll
        for (int i = 0; i < 2; ++i) {
            #pragma unroll
            for (int r = 0; r < 8; ++r) {
                const int t  = m0 + i * 32 + mh + r;
                const int bb = t >> 11;        // batch
                const int s  = t & (S_ - 1);   // seq pos
                float val = c[i][j][r] + bv;
                if (n < D_) {                  // Q (pre-scaled by 1/sqrt(hd))
                    const int h = n >> 6, d = n & 63;
                    qbuf[((((size_t)bb * H_ + h) * S_ + s) << 6) + d] =
                        (_Float16)(val * QK_SCALE);
                } else if (n < 2 * D_) {       // K
                    const int nn = n - D_, h = nn >> 6, d = nn & 63;
                    kbuf[((((size_t)bb * H_ + h) * S_ + s) << 6) + d] = (_Float16)val;
                } else {                       // V, transposed [hd][S]
                    const int nn = n - 2 * D_, h = nn >> 6, d = nn & 63;
                    vtbuf[(((size_t)bb * H_ + h) * HD_ + d) * S_ + s] = (_Float16)val;
                }
            }
        }
    }
}

// ---------------------------------------------------------------------------
// Kernel 2: causal flash attention. One wave per (b, h, 16-row q tile).
// 64-key chunks: 8 WMMAs Q@K^T, online softmax (shfl row reductions),
// P->LDS re-swizzle, 8 WMMAs P@V into 16x64 f32 accumulators.
// ---------------------------------------------------------------------------
__global__ void __launch_bounds__(32)
k_attn(const _Float16* __restrict__ qbuf,
       const _Float16* __restrict__ kbuf,
       const _Float16* __restrict__ vtbuf,
       _Float16* __restrict__ ctxbuf) {
    const int lane = threadIdx.x & 31;
    const int qt = blockIdx.x & 127;         // q tile within (b,h)
    const int bh = blockIdx.x >> 7;          // b*H + h
    const int b  = bh >> 4;
    const int h  = bh & 15;
    const int q0 = qt * 16;

    const _Float16* qhead  = qbuf  + (size_t)bh * S_ * HD_;
    const _Float16* khead  = kbuf  + (size_t)bh * S_ * HD_;
    const _Float16* vthead = vtbuf + (size_t)bh * HD_ * S_;

    // Q fragments for this tile (K dim = hd = 64 -> two 16x32 A fragments)
    const v16h aq0 = load_a16x32(qhead + (size_t)q0 * HD_,      HD_, lane);
    const v16h aq1 = load_a16x32(qhead + (size_t)q0 * HD_ + 32, HD_, lane);

    v8f ctx[4] = {};
    float mrow[8], lrow[8];
    #pragma unroll
    for (int r = 0; r < 8; ++r) { mrow[r] = -1e30f; lrow[r] = 0.0f; }

    __shared__ _Float16 pTile[16 * 64];      // P chunk, row-major 16x64

    const int rowBase = q0 + ((lane >> 4) * 8);
    const int nkv = q0 + 16;                 // causal limit (exclusive)

    for (int kv0 = 0; kv0 < nkv; kv0 += 64) {
        // ---- scores: S = Q(16x64) @ K^T(64x64), four 16-key N tiles ----
        v8f sc[4] = {};
        #pragma unroll
        for (int j = 0; j < 4; ++j) {
            if (kv0 + j * 16 < nkv) {        // skip fully-masked key subtiles
                const _Float16* kb = khead + (size_t)(kv0 + j * 16) * HD_;
                sc[j] = wmma_f16(aq0, load_b32x16(kb,      HD_, lane), sc[j]);
                sc[j] = wmma_f16(aq1, load_b32x16(kb + 32, HD_, lane), sc[j]);
            }
        }

        // ---- online softmax per row r (row spread over 16 lanes) ----
        #pragma unroll
        for (int r = 0; r < 8; ++r) {
            const int row = rowBase + r;
            float v[4];
            #pragma unroll
            for (int j = 0; j < 4; ++j) {
                const int col = kv0 + j * 16 + (lane & 15);
                v[j] = (col <= row) ? sc[j][r] : -1e30f;
            }

            float cm = fmaxf(fmaxf(v[0], v[1]), fmaxf(v[2], v[3]));
            #pragma unroll
            for (int off = 8; off > 0; off >>= 1)
                cm = fmaxf(cm, __shfl_xor(cm, off, 32));

            const float newm  = fmaxf(mrow[r], cm);
            const float alpha = __expf(mrow[r] - newm);
            mrow[r] = newm;

            float psum = 0.0f;
            const int rloc = ((lane >> 4) * 8) + r;
            #pragma unroll
            for (int j = 0; j < 4; ++j) {
                const float p = __expf(v[j] - newm);   // masked -> 0
                psum += p;
                pTile[rloc * 64 + j * 16 + (lane & 15)] = (_Float16)p;
            }
            #pragma unroll
            for (int off = 8; off > 0; off >>= 1)
                psum += __shfl_xor(psum, off, 32);

            lrow[r] = lrow[r] * alpha + psum;
            ctx[0][r] *= alpha; ctx[1][r] *= alpha;
            ctx[2][r] *= alpha; ctx[3][r] *= alpha;
        }

        // re-swizzle P (C-layout -> A-layout) through LDS: two 16x32 A frags
        const v16h pa0 = load_a16x32(pTile,      64, lane);   // keys kv0..kv0+31
        const v16h pa1 = load_a16x32(pTile + 32, 64, lane);   // keys kv0+32..kv0+63
        const bool hiHalf = (kv0 + 32) < nkv;                 // upper 32 keys live?

        // ---- ctx += P(16x64) @ V(64x64): 4 hd-column N tiles ----
        #pragma unroll
        for (int j = 0; j < 4; ++j) {
            const _Float16* vb = vthead + (size_t)(j * 16) * S_ + kv0;
            ctx[j] = wmma_f16(pa0, load_b32x16(vb, S_, lane), ctx[j]);
            if (hiHalf)
                ctx[j] = wmma_f16(pa1, load_b32x16(vb + 32, S_, lane), ctx[j]);
        }
    }

    // ---- epilogue: normalize by row sums, write ctx as f16 [B,S,D] ----
    const int ncol = lane & 15;
    #pragma unroll
    for (int r = 0; r < 8; ++r) {
        const int s = rowBase + r;
        const float inv = 1.0f / lrow[r];
        _Float16* dst = ctxbuf + ((size_t)b * S_ + s) * D_ + h * HD_ + ncol;
        dst[0]  = (_Float16)(ctx[0][r] * inv);
        dst[16] = (_Float16)(ctx[1][r] * inv);
        dst[32] = (_Float16)(ctx[2][r] * inv);
        dst[48] = (_Float16)(ctx[3][r] * inv);
    }
}

// ---------------------------------------------------------------------------
// Kernel 3: out projection + bias + residual:
//   yres[t][n] = sum_k ctx[t][k]*Wo[n][k] + ob[n] + x[t][n]
// ---------------------------------------------------------------------------
__global__ void k_oproj_gemm(const _Float16* __restrict__ ctxh,
                             const _Float16* __restrict__ woh,
                             const float* __restrict__ ob,
                             const float* __restrict__ xres,
                             float* __restrict__ yres) {
    const int lane = threadIdx.x & 31;
    const int wave = threadIdx.x >> 5;
    const int m0 = blockIdx.x * 64 + (wave >> 2) * 32;
    const int n0 = blockIdx.y * 256 + (wave & 3) * 64;

    v8f c[2][4] = {};
    gemm_core_32x64(ctxh + (size_t)m0 * D_, woh + (size_t)n0 * D_, lane, c);

    const int mh = (lane >> 4) * 8;
    #pragma unroll
    for (int j = 0; j < 4; ++j) {
        const int n  = n0 + j * 16 + (lane & 15);
        const float bv = ob[n];
        #pragma unroll
        for (int i = 0; i < 2; ++i) {
            #pragma unroll
            for (int r = 0; r < 8; ++r) {
                const size_t t = (size_t)(m0 + i * 32 + mh + r);
                yres[t * D_ + n] = c[i][j][r] + bv + xres[t * D_ + n];
            }
        }
    }
}

// ---------------------------------------------------------------------------
// Kernel 4: LayerNorm over D=1024 per token row
// ---------------------------------------------------------------------------
__global__ void k_layernorm(const float* __restrict__ yres,
                            const float* __restrict__ gamma,
                            const float* __restrict__ beta,
                            float* __restrict__ out) {
    const int row = blockIdx.x;
    const int tid = threadIdx.x;
    const float* y = yres + (size_t)row * D_;

    float v[4], s = 0.0f, s2 = 0.0f;
    #pragma unroll
    for (int j = 0; j < 4; ++j) {
        v[j] = y[tid + j * 256];
        s  += v[j];
        s2 += v[j] * v[j];
    }

    __shared__ float rs[256], rq[256];
    rs[tid] = s; rq[tid] = s2;
    __syncthreads();
    for (int off = 128; off > 0; off >>= 1) {
        if (tid < off) { rs[tid] += rs[tid + off]; rq[tid] += rq[tid + off]; }
        __syncthreads();
    }
    const float mu  = rs[0] * (1.0f / D_);
    const float var = rq[0] * (1.0f / D_) - mu * mu;
    const float inv = rsqrtf(var + 1e-5f);

    float* o = out + (size_t)row * D_;
    #pragma unroll
    for (int j = 0; j < 4; ++j) {
        const int c = tid + j * 256;
        o[c] = (v[j] - mu) * inv * gamma[c] + beta[c];
    }
}

// ---------------------------------------------------------------------------
// Host launcher
// ---------------------------------------------------------------------------
extern "C" void kernel_launch(void* const* d_in, const int* in_sizes, int n_in,
                              void* d_out, int out_size, void* d_ws, size_t ws_size,
                              hipStream_t stream) {
    (void)in_sizes; (void)n_in; (void)out_size; (void)ws_size;

    const float* x        = (const float*)d_in[0];   // [B,S,D]
    const float* in_w     = (const float*)d_in[1];   // [3D,D]
    const float* in_b     = (const float*)d_in[2];   // [3D]
    const float* out_w    = (const float*)d_in[3];   // [D,D]
    const float* out_b    = (const float*)d_in[4];   // [D]
    const float* gamma    = (const float*)d_in[5];   // [D]
    const float* beta     = (const float*)d_in[6];   // [D]
    float* out            = (float*)d_out;           // [B,S,D]

    // ---- workspace carve-up (all sizes multiples of 256B) ----
    char* ws = (char*)d_ws;
    _Float16* xh    = (_Float16*)ws; ws += (size_t)M_TOK * D_ * 2;      // 16 MB
    _Float16* wqkvh = (_Float16*)ws; ws += (size_t)N_QKV * D_ * 2;      // 6 MB
    _Float16* woh   = (_Float16*)ws; ws += (size_t)D_ * D_ * 2;         // 2 MB
    _Float16* qh    = (_Float16*)ws; ws += (size_t)M_TOK * D_ * 2;      // 16 MB
    _Float16* kh    = (_Float16*)ws; ws += (size_t)M_TOK * D_ * 2;      // 16 MB
    _Float16* vth   = (_Float16*)ws; ws += (size_t)M_TOK * D_ * 2;      // 16 MB
    _Float16* ctxh  = (_Float16*)ws; ws += (size_t)M_TOK * D_ * 2;      // 16 MB
    float*    yres  = (float*)ws;    ws += (size_t)M_TOK * D_ * 4;      // 32 MB

    // 0) fp32 -> fp16
    const int nx = M_TOK * D_;          // 8388608
    const int nw = N_QKV * D_;          // 3145728
    const int no = D_ * D_;             // 1048576
    k_f32_to_f16<<<(nx + 255) / 256, 256, 0, stream>>>(x, xh, nx);
    k_f32_to_f16<<<(nw + 255) / 256, 256, 0, stream>>>(in_w, wqkvh, nw);
    k_f32_to_f16<<<(no + 255) / 256, 256, 0, stream>>>(out_w, woh, no);

    // 1) QKV projection: M=8192, N=3072 -> grid (128, 12), 8 waves/block
    k_qkv_gemm<<<dim3(M_TOK / 64, N_QKV / 256), 256, 0, stream>>>(
        xh, wqkvh, in_b, qh, kh, vth);

    // 2) causal flash attention: one wave per (b,h,q-tile)
    k_attn<<<B_ * H_ * (S_ / 16), 32, 0, stream>>>(qh, kh, vth, ctxh);

    // 3) out projection + residual: M=8192, N=1024 -> grid (128, 4)
    k_oproj_gemm<<<dim3(M_TOK / 64, D_ / 256), 256, 0, stream>>>(
        ctxh, woh, out_b, x, yres);

    // 4) LayerNorm
    k_layernorm<<<M_TOK, 256, 0, stream>>>(yres, gamma, beta, out);
}
